// AssociationScore_3453153706626
// MI455X (gfx1250) — compile-verified, matrix operand order checked by reading, not run on hardware
//
#include <hip/hip_runtime.h>
#include <hip/hip_bf16.h>

#define IN_DIM_ 256
#define HIDDEN_ 64

typedef __attribute__((ext_vector_type(16))) __bf16 v16bf;
typedef __attribute__((ext_vector_type(8)))  float  v8f;

// ---------------------------------------------------------------------------
// Pack W [256 x 64] (f32, row-major) into per-lane WMMA B-fragments (bf16).
// Fragment index t = nt*256 + ks*32 + lane; each fragment = 16 bf16 (32 B).
// Layout (wave32 16-bit B 32x16): lane&15 = column, lane>>4 selects K blocks
// {0..7,16..23} vs {8..15,24..31} within the 32-wide K step.
// ---------------------------------------------------------------------------
__global__ void k_pack_bfrag(const float* __restrict__ W, __bf16* __restrict__ bfrag) {
    int t = blockIdx.x * blockDim.x + threadIdx.x;
    if (t >= 4 * 8 * 32) return;
    int lane  = t & 31;
    int ks    = (t >> 5) & 7;
    int nt    = t >> 8;                 // 0..3 : 16-col tile
    int col   = nt * 16 + (lane & 15);
    int khalf = lane >> 4;
    int kbase = ks * 32 + khalf * 8;
    __bf16* dst = bfrag + t * 16;
#pragma unroll
    for (int i = 0; i < 8; ++i) {
        dst[i]     = (__bf16)W[(kbase + i)      * HIDDEN_ + col];
        dst[8 + i] = (__bf16)W[(kbase + 16 + i) * HIDDEN_ + col];
    }
}

// ---------------------------------------------------------------------------
// h = x @ W via v_wmma_f32_16x16x32_bf16. One wave = one 16-row tile x all
// 64 output columns (4 x v8f accumulators). K=256 -> 8 WMMA steps per tile.
// ---------------------------------------------------------------------------
__global__ void __launch_bounds__(256) k_gemm_wmma(
        const float* __restrict__ x, const __bf16* __restrict__ bfrag,
        float* __restrict__ h, int nTiles) {
    int wave = threadIdx.x >> 5;
    int lane = threadIdx.x & 31;
    int tile = blockIdx.x * 8 + wave;
    if (tile >= nTiles) return;                 // wave-uniform exit (EXEC all-1 for WMMA)
    int rowBase = tile * 16;
    int khalf   = lane >> 4;
    const float* xrow = x + (size_t)(rowBase + (lane & 15)) * IN_DIM_ + khalf * 8;
    const v16bf* bp = (const v16bf*)bfrag;

    v8f acc0 = {}, acc1 = {}, acc2 = {}, acc3 = {};

#pragma unroll
    for (int ks = 0; ks < 8; ++ks) {
        const float* p = xrow + ks * 32;
        float4 f0 = *(const float4*)(p);
        float4 f1 = *(const float4*)(p + 4);
        float4 f2 = *(const float4*)(p + 16);
        float4 f3 = *(const float4*)(p + 20);
        if (ks < 7) __builtin_prefetch(p + 32, 0, 1);   // global_prefetch_b8

        v16bf a;
        a[0]  = (__bf16)f0.x; a[1]  = (__bf16)f0.y; a[2]  = (__bf16)f0.z; a[3]  = (__bf16)f0.w;
        a[4]  = (__bf16)f1.x; a[5]  = (__bf16)f1.y; a[6]  = (__bf16)f1.z; a[7]  = (__bf16)f1.w;
        a[8]  = (__bf16)f2.x; a[9]  = (__bf16)f2.y; a[10] = (__bf16)f2.z; a[11] = (__bf16)f2.w;
        a[12] = (__bf16)f3.x; a[13] = (__bf16)f3.y; a[14] = (__bf16)f3.z; a[15] = (__bf16)f3.w;

        v16bf b0 = bp[(0 * 8 + ks) * 32 + lane];
        v16bf b1 = bp[(1 * 8 + ks) * 32 + lane];
        v16bf b2v = bp[(2 * 8 + ks) * 32 + lane];
        v16bf b3 = bp[(3 * 8 + ks) * 32 + lane];
        acc0 = __builtin_amdgcn_wmma_f32_16x16x32_bf16(false, a, false, b0, (short)0, acc0, false, false);
        acc1 = __builtin_amdgcn_wmma_f32_16x16x32_bf16(false, a, false, b1, (short)0, acc1, false, false);
        acc2 = __builtin_amdgcn_wmma_f32_16x16x32_bf16(false, a, false, b2v, (short)0, acc2, false, false);
        acc3 = __builtin_amdgcn_wmma_f32_16x16x32_bf16(false, a, false, b3, (short)0, acc3, false, false);
    }

    // C/D layout: VGPR r -> M = r + 8*(lane>=16), N = lane&15
    int m0 = khalf * 8;
    int n  = lane & 15;
    float* hout = h + (size_t)rowBase * HIDDEN_ + n;
#pragma unroll
    for (int r = 0; r < 8; ++r) {
        float* hr = hout + (size_t)(m0 + r) * HIDDEN_;
        hr[0]  = acc0[r];
        hr[16] = acc1[r];
        hr[32] = acc2[r];
        hr[48] = acc3[r];
    }
}

// Scalar tail for N % 16 != 0 (not hit for N=100000, kept for generality).
__global__ void k_gemm_tail(const float* __restrict__ x, const float* __restrict__ W,
                            float* __restrict__ h, int rowStart, int N) {
    int idx = blockIdx.x * blockDim.x + threadIdx.x;
    int row = rowStart + idx / HIDDEN_;
    int col = idx % HIDDEN_;
    if (row >= N) return;
    float s = 0.f;
    for (int k = 0; k < IN_DIM_; ++k) s += x[(size_t)row * IN_DIM_ + k] * W[k * HIDDEN_ + col];
    h[(size_t)row * HIDDEN_ + col] = s;
}

// ---------------------------------------------------------------------------
// Degree / norm kernels
// ---------------------------------------------------------------------------
__global__ void k_deg_init(float* __restrict__ deg, int N) {
    int i = blockIdx.x * blockDim.x + threadIdx.x;
    if (i < N) deg[i] = 1.0f;                   // self-loop contribution
}

__global__ void k_deg_count(const int* __restrict__ ei, int E, float* __restrict__ deg) {
    int e = blockIdx.x * blockDim.x + threadIdx.x;
    if (e < E) atomicAdd(&deg[ei[E + e]], 1.0f);   // dst row of edge_index
}

__global__ void k_dinv(const float* __restrict__ deg, float* __restrict__ dinv, int N) {
    int i = blockIdx.x * blockDim.x + threadIdx.x;
    if (i < N) {
        float d = deg[i];
        dinv[i] = d > 0.f ? rsqrtf(d) : 0.f;
    }
}

// agg[i] = dinv[i]^2 * h[i]   (self-loop term; edges added on top)
__global__ void k_agg_init(const float* __restrict__ h, const float* __restrict__ dinv,
                           float* __restrict__ agg, int total) {
    int i = blockIdx.x * blockDim.x + threadIdx.x;
    if (i >= total) return;
    float di = dinv[i >> 6];
    agg[i] = di * di * h[i];
}

// Edge scatter: thread = (edge, feat), feat fastest -> coalesced row reads and
// same-cacheline f32 atomics. h and agg are L2-resident (25.6 MB each).
__global__ void k_scatter(const int* __restrict__ ei, int E,
                          const float* __restrict__ dinv,
                          const float* __restrict__ h,
                          float* __restrict__ agg, int total) {
    int gid = blockIdx.x * blockDim.x + threadIdx.x;
    if (gid >= total) return;
    int e = gid >> 6;
    int f = gid & 63;
    int s = ei[e];
    int d = ei[E + e];
    float nrm = dinv[s] * dinv[d];
    atomicAdd(&agg[(size_t)d * HIDDEN_ + f], nrm * h[(size_t)s * HIDDEN_ + f]);
}

// score = sigmoid((agg + b) . w2 + b2)
__global__ void k_score(const float* __restrict__ agg, const float* __restrict__ b,
                        const float* __restrict__ w2, const float* __restrict__ b2,
                        float* __restrict__ out, int N) {
    int i = blockIdx.x * blockDim.x + threadIdx.x;
    if (i >= N) return;
    const float* ag = agg + (size_t)i * HIDDEN_;
    float s = b2[0];
#pragma unroll
    for (int f = 0; f < HIDDEN_; ++f) s += (ag[f] + b[f]) * w2[f];
    out[i] = 1.0f / (1.0f + __expf(-s));
}

// ---------------------------------------------------------------------------
extern "C" void kernel_launch(void* const* d_in, const int* in_sizes, int n_in,
                              void* d_out, int out_size, void* d_ws, size_t ws_size,
                              hipStream_t stream) {
    const float* x  = (const float*)d_in[0];
    const int*   ei = (const int*)d_in[1];
    const float* W  = (const float*)d_in[2];
    const float* b  = (const float*)d_in[3];
    const float* w2 = (const float*)d_in[4];
    const float* b2 = (const float*)d_in[5];
    float* out = (float*)d_out;

    const int N = in_sizes[0] / IN_DIM_;   // 100000
    const int E = in_sizes[1] / 2;         // 1600000

    // Workspace carve-up
    float*  h     = (float*)d_ws;                          // N*64 f32
    float*  agg   = h + (size_t)N * HIDDEN_;               // N*64 f32
    float*  deg   = agg + (size_t)N * HIDDEN_;             // N f32
    float*  dinv  = deg + N;                               // N f32
    __bf16* bfrag = (__bf16*)(dinv + N);                   // 4*8*32*16 bf16 (32 KB)

    k_pack_bfrag<<<4, 256, 0, stream>>>(W, bfrag);
    k_deg_init  <<<(N + 255) / 256, 256, 0, stream>>>(deg, N);
    k_deg_count <<<(E + 255) / 256, 256, 0, stream>>>(ei, E, deg);
    k_dinv      <<<(N + 255) / 256, 256, 0, stream>>>(deg, dinv, N);

    int nTiles = N / 16;
    k_gemm_wmma<<<(nTiles + 7) / 8, 256, 0, stream>>>(x, bfrag, h, nTiles);
    int rem = N - nTiles * 16;
    if (rem > 0) {
        int tailThreads = rem * HIDDEN_;
        k_gemm_tail<<<(tailThreads + 255) / 256, 256, 0, stream>>>(x, W, h, nTiles * 16, N);
    }

    int aggTotal = N * HIDDEN_;
    k_agg_init<<<(aggTotal + 255) / 256, 256, 0, stream>>>(h, dinv, agg, aggTotal);

    int scatTotal = E * HIDDEN_;           // 102.4M threads
    k_scatter<<<(scatTotal + 255) / 256, 256, 0, stream>>>(ei, E, dinv, h, agg, scatTotal);

    k_score<<<(N + 255) / 256, 256, 0, stream>>>(agg, b, w2, b2, out, N);
}
